// MultiheadSelfAttention_57234734186528
// MI455X (gfx1250) — matrix operand order repaired and checked by reading, not compiled, not run on hardware
//
#include <hip/hip_runtime.h>
#include <hip/hip_bf16.h>

// ---- problem constants -----------------------------------------------------
#define D_IN  1024
#define HD    1024
#define NH    16
#define HDIM  64
#define SEQ   2048
#define BS    2
#define ROWS  (BS * SEQ)      // 4096

typedef __attribute__((ext_vector_type(16))) _Float16 v16h;
typedef __attribute__((ext_vector_type(8)))  float    v8f;

// ---- WMMA helpers ----------------------------------------------------------
__device__ __forceinline__ v8f wmma16(v16h a, v16h b, v8f c) {
  // D = A(16x32 f16) x B(32x16 f16) + C(16x16 f32)
  return __builtin_amdgcn_wmma_f32_16x16x32_f16(
      /*neg_a=*/false, a, /*neg_b=*/false, b,
      /*c_mod=*/(short)0, c, /*reuse_a=*/false, /*reuse_b=*/false);
}

// A-matrix 16x32 fragment: lane<16 -> M=lane, K=0..15 ; lane>=16 -> K=16..31
__device__ __forceinline__ v16h ld_fragA(const _Float16* p) {
  v16h h;
  __builtin_memcpy(reinterpret_cast<char*>(&h) + 0,  p,      16);
  __builtin_memcpy(reinterpret_cast<char*>(&h) + 16, p + 8, 16);
  return h;
}
// B-matrix 32x16 fragment: lane = (n&15) + 16*(k>>4); 16 contiguous K halves per lane.
__device__ __forceinline__ v16h ld_fragB(const _Float16* p) {
  v16h h;
  __builtin_memcpy(reinterpret_cast<char*>(&h) + 0,  p,     16);
  __builtin_memcpy(reinterpret_cast<char*>(&h) + 16, p + 8, 16);
  return h;
}

// ---- CDNA5 async global->LDS copy (ASYNCcnt path, §15.18.3 op 98) ----------
// Copies 32 contiguous bytes: INST_OFFSET is added to BOTH the LDS and the
// global address, so two b128 ops with offset:0 / offset:16 move the block.
__device__ __forceinline__ void async_copy32(unsigned lds_addr, const _Float16* g) {
  asm volatile("global_load_async_to_lds_b128 %0, %1, off"
               :: "v"(lds_addr), "v"(g) : "memory");
  asm volatile("global_load_async_to_lds_b128 %0, %1, off offset:16"
               :: "v"(lds_addr), "v"(g) : "memory");
}
__device__ __forceinline__ void wait_async0() {
  asm volatile("s_wait_asynccnt 0x0" ::: "memory");
}

// ---- cast kernels ----------------------------------------------------------
__global__ __launch_bounds__(256) void cast_f16_kernel(
    const float* __restrict__ s, _Float16* __restrict__ d, int n) {
  int i = (blockIdx.x * 256 + threadIdx.x) * 4;
  if (i >= n) return;
  float4 v;
  __builtin_memcpy(&v, s + i, 16);
  _Float16 h[4];
  h[0] = (_Float16)v.x; h[1] = (_Float16)v.y; h[2] = (_Float16)v.z; h[3] = (_Float16)v.w;
  __builtin_memcpy(d + i, h, 8);
}

// W (K x N) f32 -> Wt (N x K) f16, tiled 32x32 through LDS
__global__ __launch_bounds__(256) void transpose_cast_kernel(
    const float* __restrict__ W, _Float16* __restrict__ Wt) {
  __shared__ float tile[32][33];
  const int tx = threadIdx.x & 31, ty = threadIdx.x >> 5;
  const int kb = blockIdx.y * 32, nb = blockIdx.x * 32;
#pragma unroll
  for (int j = 0; j < 4; j++) {
    int k = ty + j * 8;
    tile[k][tx] = W[(size_t)(kb + k) * HD + nb + tx];
  }
  __syncthreads();
#pragma unroll
  for (int j = 0; j < 4; j++) {
    int n = ty + j * 8;
    Wt[(size_t)(nb + n) * D_IN + kb + tx] = (_Float16)tile[tx][n];
  }
}

// ---- QKV projection GEMM ---------------------------------------------------
// X(4096x1024)f16 @ W^T-layout(1024n x 1024k)f16 + bias -> (B,H,S,64) f16
// Double-buffered LDS, tiles DMA'd by GLOBAL_LOAD_ASYNC_TO_LDS.
#define TK   32
#define ASTR 40   // LDS row stride in halves (80B = 5*16B -> aligned b128)

__global__ __launch_bounds__(256) void gemm_qkv_kernel(
    const _Float16* __restrict__ Xh,
    const _Float16* __restrict__ Wt3,     // 3 stacked transposed weights
    const float* __restrict__ bq, const float* __restrict__ bk,
    const float* __restrict__ bv,
    _Float16* __restrict__ Out3)          // 3 stacked (B,H,S,64)
{
  const int z = blockIdx.z;
  const _Float16* W = Wt3 + (size_t)z * (D_IN * HD);
  _Float16* O = Out3 + (size_t)z * ((size_t)ROWS * HD);
  const float* bias = (z == 0) ? bq : (z == 1) ? bk : bv;

  __shared__ _Float16 As[2][128 * ASTR];
  __shared__ _Float16 Bs[2][128 * ASTR];

  const int t = threadIdx.x;
  const int w = t >> 5, lane = t & 31;
  const int nl = lane & 15, hi = lane >> 4;
  const int rowBase = blockIdx.y * 128;
  const int colBase = blockIdx.x * 128;
  const int wm = (w >> 1) * 32;           // wave M offset (4 x 32)
  const int wn = (w & 1) * 64;            // wave N offset (2 x 64)
  const int lr = t >> 1, lc = (t & 1) * 16;

  const _Float16* gA = Xh + (size_t)(rowBase + lr) * D_IN + lc;
  const _Float16* gB = W  + (size_t)(colBase + lr) * D_IN + lc;
  const unsigned lA0 = (unsigned)(size_t)&As[0][lr * ASTR + lc];
  const unsigned lA1 = (unsigned)(size_t)&As[1][lr * ASTR + lc];
  const unsigned lB0 = (unsigned)(size_t)&Bs[0][lr * ASTR + lc];
  const unsigned lB1 = (unsigned)(size_t)&Bs[1][lr * ASTR + lc];

  v8f zero = {};
  v8f acc[2][4];
#pragma unroll
  for (int i = 0; i < 2; i++)
#pragma unroll
    for (int j = 0; j < 4; j++) acc[i][j] = zero;

  // prologue: DMA tile 0 into buffer 0
  async_copy32(lA0, gA);
  async_copy32(lB0, gB);
  wait_async0();
  __syncthreads();

  const int NT = D_IN / TK;               // 32 K-steps
  for (int i = 0; i < NT; i++) {
    const int cur = i & 1;
    if (i + 1 < NT) {                     // DMA tile i+1 into the other buffer
      async_copy32(cur ? lA0 : lA1, gA + (size_t)(i + 1) * TK);
      async_copy32(cur ? lB0 : lB1, gB + (size_t)(i + 1) * TK);
    }
    const _Float16* Ac = As[cur];
    const _Float16* Bc = Bs[cur];
    v16h af[2], bf[4];
#pragma unroll
    for (int mi = 0; mi < 2; mi++)
      af[mi] = ld_fragA(&Ac[(wm + mi * 16 + nl) * ASTR + hi * 16]);
#pragma unroll
    for (int ni = 0; ni < 4; ni++)
      bf[ni] = ld_fragB(&Bc[(wn + ni * 16 + nl) * ASTR + hi * 16]);
#pragma unroll
    for (int mi = 0; mi < 2; mi++)
#pragma unroll
      for (int ni = 0; ni < 4; ni++)
        acc[mi][ni] = wmma16(af[mi], bf[ni], acc[mi][ni]);
    wait_async0();                        // next tile resident
    __syncthreads();                      // everyone done reading cur / sees next
  }

  // epilogue: +bias, scatter into (B,H,S,64)
#pragma unroll
  for (int mi = 0; mi < 2; mi++)
#pragma unroll
    for (int ni = 0; ni < 4; ni++) {
      const int col = colBase + wn + ni * 16 + nl;
      const int hh = col >> 6, dd = col & 63;
      const float bsv = bias[col];
#pragma unroll
      for (int r = 0; r < 8; r++) {
        const int row = rowBase + wm + mi * 16 + r + 8 * hi;
        const int bb = row >> 11, ss = row & (SEQ - 1);
        O[((size_t)(bb * NH + hh) * SEQ + ss) * HDIM + dd] =
            (_Float16)(acc[mi][ni][r] + bsv);
      }
    }
}

// ---- flash attention -------------------------------------------------------
#define VT_S 136
#define PB_S 136

__global__ __launch_bounds__(256) void attn_kernel(
    const _Float16* __restrict__ Qh, const _Float16* __restrict__ Kh,
    const _Float16* __restrict__ Vh, _Float16* __restrict__ Ctx) {
  __shared__ _Float16 Vt[HDIM * VT_S];    // V^T tile: [d][key]
  __shared__ _Float16 Pb[128 * PB_S];     // per-wave P buffers (8 x 16 x 128)

  const int t = threadIdx.x;
  const int w = t >> 5, lane = t & 31;
  const int nl = lane & 15, hi = lane >> 4;
  const int qt = blockIdx.x, h = blockIdx.y, b = blockIdx.z;
  const size_t headOff = (size_t)(b * NH + h) * SEQ * HDIM;
  const _Float16* Qp = Qh + headOff;
  const _Float16* Kp = Kh + headOff;
  const _Float16* Vp = Vh + headOff;
  const int qrow = qt * 128 + w * 16;     // this wave's 16 query rows
  _Float16* Pw = Pb + w * 16 * PB_S;

  // Q A-fragments straight from global (lane layout is 16B-contiguous)
  v16h qf[2];
#pragma unroll
  for (int kc = 0; kc < 2; kc++)
    qf[kc] = ld_fragA(Qp + (size_t)(qrow + nl) * HDIM + kc * 32 + hi * 16);

  v8f zero = {};
  v8f oacc[4];
#pragma unroll
  for (int nb = 0; nb < 4; nb++) oacc[nb] = zero;
  float rmax[8], rsum[8];
#pragma unroll
  for (int r = 0; r < 8; r++) { rmax[r] = -1e30f; rsum[r] = 0.f; }

  for (int kt = 0; kt < SEQ; kt += 128) {
    __syncthreads();
    {   // cooperative V tile load + transpose into LDS: Vt[d][key]
      const int key = t >> 1;
      const int dbase = (t & 1) * 32;
      _Float16 tv[32];
      __builtin_memcpy(tv, Vp + (size_t)(kt + key) * HDIM + dbase, 64);
#pragma unroll
      for (int i = 0; i < 32; i++) Vt[(dbase + i) * VT_S + key] = tv[i];
    }
    // high-locality prefetch of next K tile (WGP-scope path)
    if (kt + 128 < SEQ)
      __builtin_prefetch(Kp + (size_t)(kt + 128 + (t >> 1)) * HDIM + (t & 1) * 32, 0, 3);
    __syncthreads();

    // S = Q @ K^T : 16 rows x 128 keys in 8 accumulators
    v8f sc[8];
#pragma unroll
    for (int nb = 0; nb < 8; nb++) {
      v8f c = zero;
#pragma unroll
      for (int kc = 0; kc < 2; kc++) {
        v16h kf = ld_fragB(Kp + (size_t)(kt + nb * 16 + nl) * HDIM + kc * 32 + hi * 16);
        c = wmma16(qf[kc], kf, c);
      }
      sc[nb] = c;
    }
#pragma unroll
    for (int nb = 0; nb < 8; nb++) sc[nb] *= 0.125f;   // 1/sqrt(64)

    // online softmax: row m = r + 8*hi lives on 16 lanes of one half
    float fac[8];
#pragma unroll
    for (int r = 0; r < 8; r++) {
      float m = sc[0][r];
#pragma unroll
      for (int nb = 1; nb < 8; nb++) m = fmaxf(m, sc[nb][r]);
      m = fmaxf(m, __shfl_xor(m, 1, 32));
      m = fmaxf(m, __shfl_xor(m, 2, 32));
      m = fmaxf(m, __shfl_xor(m, 4, 32));
      m = fmaxf(m, __shfl_xor(m, 8, 32));
      float nm = fmaxf(rmax[r], m);
      fac[r] = __expf(rmax[r] - nm);
      rmax[r] = nm;
    }
#pragma unroll
    for (int nb = 0; nb < 4; nb++)
#pragma unroll
      for (int r = 0; r < 8; r++) oacc[nb][r] *= fac[r];

    // P = exp(S - max); row sums; spill P (f16) to per-wave LDS in row-major
#pragma unroll
    for (int r = 0; r < 8; r++) {
      float s = 0.f;
      const int mr = r + 8 * hi;
#pragma unroll
      for (int nb = 0; nb < 8; nb++) {
        float p = __expf(sc[nb][r] - rmax[r]);
        s += p;
        Pw[mr * PB_S + nb * 16 + nl] = (_Float16)p;
      }
      s += __shfl_xor(s, 1, 32);
      s += __shfl_xor(s, 2, 32);
      s += __shfl_xor(s, 4, 32);
      s += __shfl_xor(s, 8, 32);
      rsum[r] = rsum[r] * fac[r] + s;
    }

    // O += P(16x128) @ V(128x64)
#pragma unroll
    for (int kc = 0; kc < 4; kc++) {
      v16h pf = ld_fragA(&Pw[nl * PB_S + kc * 32 + hi * 16]);
#pragma unroll
      for (int nb = 0; nb < 4; nb++) {
        v16h vf = ld_fragB(&Vt[(nb * 16 + nl) * VT_S + kc * 32 + hi * 16]);
        oacc[nb] = wmma16(pf, vf, oacc[nb]);
      }
    }
  }

  float inv[8];
#pragma unroll
  for (int r = 0; r < 8; r++) inv[r] = 1.0f / rsum[r];
#pragma unroll
  for (int nb = 0; nb < 4; nb++)
#pragma unroll
    for (int r = 0; r < 8; r++) {
      const int row = b * SEQ + qrow + r + 8 * hi;
      const int col = h * HDIM + nb * 16 + nl;
      Ctx[(size_t)row * HD + col] = (_Float16)(oacc[nb][r] * inv[r]);
    }
}

// ---- output projection GEMM (+bias +residual, f32 out) ---------------------
__global__ __launch_bounds__(256) void gemm_out_kernel(
    const _Float16* __restrict__ Ah,      // ctx 4096x1024 f16
    const _Float16* __restrict__ Wt,      // Wo transposed, f16
    const float* __restrict__ bo,
    const float* __restrict__ Xf,         // residual (f32)
    float* __restrict__ Y) {
  __shared__ _Float16 As[2][128 * ASTR];
  __shared__ _Float16 Bs[2][128 * ASTR];

  const int t = threadIdx.x;
  const int w = t >> 5, lane = t & 31;
  const int nl = lane & 15, hi = lane >> 4;
  const int rowBase = blockIdx.y * 128;
  const int colBase = blockIdx.x * 128;
  const int wm = (w >> 1) * 32;
  const int wn = (w & 1) * 64;
  const int lr = t >> 1, lc = (t & 1) * 16;

  const _Float16* gA = Ah + (size_t)(rowBase + lr) * HD + lc;
  const _Float16* gB = Wt + (size_t)(colBase + lr) * HD + lc;
  const unsigned lA0 = (unsigned)(size_t)&As[0][lr * ASTR + lc];
  const unsigned lA1 = (unsigned)(size_t)&As[1][lr * ASTR + lc];
  const unsigned lB0 = (unsigned)(size_t)&Bs[0][lr * ASTR + lc];
  const unsigned lB1 = (unsigned)(size_t)&Bs[1][lr * ASTR + lc];

  v8f zero = {};
  v8f acc[2][4];
#pragma unroll
  for (int i = 0; i < 2; i++)
#pragma unroll
    for (int j = 0; j < 4; j++) acc[i][j] = zero;

  async_copy32(lA0, gA);
  async_copy32(lB0, gB);
  wait_async0();
  __syncthreads();

  const int NT = HD / TK;
  for (int i = 0; i < NT; i++) {
    const int cur = i & 1;
    if (i + 1 < NT) {
      async_copy32(cur ? lA0 : lA1, gA + (size_t)(i + 1) * TK);
      async_copy32(cur ? lB0 : lB1, gB + (size_t)(i + 1) * TK);
    }
    const _Float16* Ac = As[cur];
    const _Float16* Bc = Bs[cur];
    v16h af[2], bf[4];
#pragma unroll
    for (int mi = 0; mi < 2; mi++)
      af[mi] = ld_fragA(&Ac[(wm + mi * 16 + nl) * ASTR + hi * 16]);
#pragma unroll
    for (int ni = 0; ni < 4; ni++)
      bf[ni] = ld_fragB(&Bc[(wn + ni * 16 + nl) * ASTR + hi * 16]);
#pragma unroll
    for (int mi = 0; mi < 2; mi++)
#pragma unroll
      for (int ni = 0; ni < 4; ni++)
        acc[mi][ni] = wmma16(af[mi], bf[ni], acc[mi][ni]);
    wait_async0();
    __syncthreads();
  }

#pragma unroll
  for (int mi = 0; mi < 2; mi++)
#pragma unroll
    for (int ni = 0; ni < 4; ni++) {
      const int col = colBase + wn + ni * 16 + nl;
      const float bsv = bo[col];
#pragma unroll
      for (int r = 0; r < 8; r++) {
        const int row = rowBase + wm + mi * 16 + r + 8 * hi;
        const size_t o = (size_t)row * HD + col;
        Y[o] = acc[mi][ni][r] + bsv + Xf[o];
      }
    }
}

// ---- LayerNorm + exact GELU, in-place on d_out -----------------------------
__global__ __launch_bounds__(256) void ln_gelu_kernel(
    float* __restrict__ Y, const float* __restrict__ gamma,
    const float* __restrict__ beta) {
  const int row = blockIdx.x;
  float* y = Y + (size_t)row * D_IN;
  const int t = threadIdx.x;
  const int lane = t & 31, w = t >> 5;

  float4 v;
  __builtin_memcpy(&v, y + t * 4, 16);
  float s  = v.x + v.y + v.z + v.w;
  float s2 = v.x * v.x + v.y * v.y + v.z * v.z + v.w * v.w;
#pragma unroll
  for (int m = 1; m < 32; m <<= 1) {
    s  += __shfl_xor(s,  m, 32);
    s2 += __shfl_xor(s2, m, 32);
  }
  __shared__ float ws1[8], ws2[8];
  if (lane == 0) { ws1[w] = s; ws2[w] = s2; }
  __syncthreads();
  float ts = 0.f, ts2 = 0.f;
#pragma unroll
  for (int i = 0; i < 8; i++) { ts += ws1[i]; ts2 += ws2[i]; }
  const float mu  = ts * (1.0f / D_IN);
  const float var = ts2 * (1.0f / D_IN) - mu * mu;
  const float inv = rsqrtf(var + 1e-5f);

  float4 g, be;
  __builtin_memcpy(&g,  gamma + t * 4, 16);
  __builtin_memcpy(&be, beta  + t * 4, 16);
  float u[4] = {v.x, v.y, v.z, v.w};
  float gg[4] = {g.x, g.y, g.z, g.w};
  float bb[4] = {be.x, be.y, be.z, be.w};
  float o[4];
#pragma unroll
  for (int i = 0; i < 4; i++) {
    float n = (u[i] - mu) * inv * gg[i] + bb[i];
    o[i] = 0.5f * n * (1.0f + erff(n * 0.70710678118654752f));
  }
  __builtin_memcpy(y + t * 4, o, 16);
}

// ---- launcher --------------------------------------------------------------
extern "C" void kernel_launch(void* const* d_in, const int* in_sizes, int n_in,
                              void* d_out, int out_size, void* d_ws, size_t ws_size,
                              hipStream_t stream) {
  const float* x     = (const float*)d_in[0];
  const float* Wq    = (const float*)d_in[1];
  const float* bq    = (const float*)d_in[2];
  const float* Wk    = (const float*)d_in[3];
  const float* bk    = (const float*)d_in[4];
  const float* Wv    = (const float*)d_in[5];
  const float* bv    = (const float*)d_in[6];
  const float* Wo    = (const float*)d_in[7];
  const float* bo    = (const float*)d_in[8];
  const float* gamma = (const float*)d_in[9];
  const float* beta  = (const float*)d_in[10];
  float* out = (float*)d_out;

  // workspace layout (f16 buffers), total 48 MB
  char* ws = (char*)d_ws;
  const size_t MB = 1u << 20;
  _Float16* xh    = (_Float16*)(ws + 0 * MB);    // 4096x1024         (8 MB)
  _Float16* Wqkvt = (_Float16*)(ws + 8 * MB);    // 3 x 1024x1024     (6 MB)
  _Float16* Wot   = (_Float16*)(ws + 14 * MB);   // 1024x1024         (2 MB)
  _Float16* QKVh  = (_Float16*)(ws + 16 * MB);   // 3 x (B,H,S,64)    (24 MB)
  _Float16* ctxh  = (_Float16*)(ws + 40 * MB);   // 4096x1024         (8 MB)

  const int NX = ROWS * D_IN;                    // 4,194,304
  cast_f16_kernel<<<NX / (256 * 4), 256, 0, stream>>>(x, xh, NX);

  dim3 tg(32, 32);
  transpose_cast_kernel<<<tg, 256, 0, stream>>>(Wq, Wqkvt + 0 * (size_t)(D_IN * HD));
  transpose_cast_kernel<<<tg, 256, 0, stream>>>(Wk, Wqkvt + 1 * (size_t)(D_IN * HD));
  transpose_cast_kernel<<<tg, 256, 0, stream>>>(Wv, Wqkvt + 2 * (size_t)(D_IN * HD));
  transpose_cast_kernel<<<tg, 256, 0, stream>>>(Wo, Wot);

  gemm_qkv_kernel<<<dim3(HD / 128, ROWS / 128, 3), 256, 0, stream>>>(
      xh, Wqkvt, bq, bk, bv, QKVh);

  const size_t hsz = (size_t)ROWS * HD;          // elements per Q/K/V tensor
  attn_kernel<<<dim3(SEQ / 128, NH, BS), 256, 0, stream>>>(
      QKVh, QKVh + hsz, QKVh + 2 * hsz, ctxh);

  gemm_out_kernel<<<dim3(HD / 128, ROWS / 128, 1), 256, 0, stream>>>(
      ctxh, Wot, bo, x, out);

  ln_gelu_kernel<<<ROWS, 256, 0, stream>>>(out, gamma, beta);
}
